// VectorQuantizer_62612033241435
// MI455X (gfx1250) — matrix operand order repaired and checked by reading, not compiled, not run on hardware
//
#include <hip/hip_runtime.h>
#include <stdint.h>

// ---------------- problem constants ----------------
#define K_CODES 8192
#define DIM 512
#define N_TOK 16384          // 8 * 2048
#define DECAY 0.99f
#define COMMIT 0.25f
#define EPSF 1e-5f

// d_out layout (float elements, reference return order)
#define OFF_ZQ      0ull
#define OFF_LOSS    ((unsigned long long)N_TOK * DIM)                 // 8388608
#define OFF_IDX     (OFF_LOSS + 1ull)
#define OFF_NEWEMB  (OFF_IDX + (unsigned long long)N_TOK)
#define OFF_CS      (OFF_NEWEMB + (unsigned long long)K_CODES * DIM)
#define OFF_ES      (OFF_CS + (unsigned long long)K_CODES)

typedef __attribute__((ext_vector_type(16))) __bf16       v16bf;
typedef __attribute__((ext_vector_type(8)))  float        v8f;
typedef __attribute__((ext_vector_type(4)))  unsigned int v4u;

// ---------------- helpers ----------------
__device__ __forceinline__ uint16_t f32_to_bf16_rne(float f) {
    uint32_t u = __float_as_uint(f);
    uint32_t r = u + 0x7FFFu + ((u >> 16) & 1u);
    return (uint16_t)(r >> 16);
}
__device__ __forceinline__ float bf16_to_f32(uint16_t h) {
    return __uint_as_float(((uint32_t)h) << 16);
}

// CDNA5 async global->LDS copy (ASYNCcnt path, ISA §10.7 / §15.18.3).
// First operand = LDS byte address VGPR, second = 64-bit global address.
__device__ __forceinline__ void async_ld_b128(uint32_t lds_off, const void* gptr) {
    asm volatile("global_load_async_to_lds_b128 %0, %1, off"
                 :: "v"(lds_off), "v"((unsigned long long)(uintptr_t)gptr)
                 : "memory");
}
__device__ __forceinline__ void wait_async0() {
    asm volatile("s_wait_asynccnt 0x0" ::: "memory");
}

// ---------------- kernel 0: zero scratch + output accumulators ----------------
__global__ void vq_init(float* __restrict__ counts, float* __restrict__ esum,
                        float* __restrict__ loss_slot, float* __restrict__ n_slot) {
    size_t i = (size_t)blockIdx.x * blockDim.x + threadIdx.x;
    if (i < (size_t)K_CODES * DIM) esum[i] = 0.f;
    if (i < K_CODES) counts[i] = 0.f;
    if (i == 0) { *loss_slot = 0.f; *n_slot = 0.f; }
}

// ---------------- kernel 1: codebook -> bf16 hi/lo planes + half-norms ----------------
__global__ void vq_prep_codebook(const float* __restrict__ emb,
                                 uint16_t* __restrict__ eh, uint16_t* __restrict__ el,
                                 float* __restrict__ hnorm) {
    const int k   = blockIdx.x;
    const int tid = threadIdx.x;
    const float* row = emb + (size_t)k * DIM;
    float ssq = 0.f;
    for (int d = tid; d < DIM; d += 256) {
        float f = row[d];
        uint16_t hi = f32_to_bf16_rne(f);
        float fl = f - bf16_to_f32(hi);
        eh[(size_t)k * DIM + d] = hi;
        el[(size_t)k * DIM + d] = f32_to_bf16_rne(fl);
        ssq += f * f;
    }
    for (int off = 16; off > 0; off >>= 1) ssq += __shfl_down(ssq, off, 32);
    __shared__ float red[8];
    if ((tid & 31) == 0) red[tid >> 5] = ssq;
    __syncthreads();
    if (tid == 0) {
        float s = 0.f;
        for (int i = 0; i < 8; ++i) s += red[i];
        hnorm[k] = 0.5f * s;
    }
}

// Async-copy one K-half (256 halves wide) of a 16-row E tile into LDS.
// 512 b128 transfers -> 4 per thread at blockDim=128.
__device__ __forceinline__ void copy_half_tile_async(const uint16_t* __restrict__ g,
                                                     uint32_t lds_base_off,
                                                     int ct, int kbase, int tid) {
    for (int i = tid; i < 512; i += 128) {
        const int row = i >> 5;
        const int col = (i & 31) << 3;           // halves
        const size_t goff = (size_t)(ct * 16 + row) * DIM + kbase + col;
        const uint32_t loff = lds_base_off + (uint32_t)(row * DIM + kbase + col) * 2u;
        async_ld_b128(loff, (const void*)(g + goff));
    }
}

// 8 K-slabs (K = kbase .. kbase+255) of the split-bf16 GEMM into 3 independent chains.
__device__ __forceinline__ void compute_half(const uint16_t* __restrict__ eh,
                                             const uint16_t* __restrict__ el,
                                             const uint16_t* __restrict__ zh,
                                             const uint16_t* __restrict__ zl,
                                             int mn, int hlf, int tokrow, int kbase,
                                             v8f& a0, v8f& a1, v8f& a2) {
    #pragma unroll
    for (int s = 0; s < 8; ++s) {
        const int k0 = kbase + s * 32;
        union { v4u q[2]; v16bf v; } ah, al, bh, bl;
        // A (E tile): lane = row mn; K = k0 + 8*hlf + {0..7} and +16 (ISA 16-bit A layout)
        const uint16_t* pa  = eh + mn * DIM + k0 + 8 * hlf;
        ah.q[0] = *(const v4u*)(pa);
        ah.q[1] = *(const v4u*)(pa + 16);
        const uint16_t* pal = el + mn * DIM + k0 + 8 * hlf;
        al.q[0] = *(const v4u*)(pal);
        al.q[1] = *(const v4u*)(pal + 16);
        // B (z tile): lane = col mn; K = k0 + 8*hlf + {0..7} and +16 (same order as A)
        const uint16_t* pb  = zh + tokrow * DIM + k0 + 8 * hlf;
        bh.q[0] = *(const v4u*)(pb);
        bh.q[1] = *(const v4u*)(pb + 16);
        const uint16_t* pbl = zl + tokrow * DIM + k0 + 8 * hlf;
        bl.q[0] = *(const v4u*)(pbl);
        bl.q[1] = *(const v4u*)(pbl + 16);
        // independent accumulator chains -> 3x WMMA-pipe ILP
        a0 = __builtin_amdgcn_wmma_f32_16x16x32_bf16(false, ah.v, false, bh.v, (short)0, a0, false, false);
        a1 = __builtin_amdgcn_wmma_f32_16x16x32_bf16(false, ah.v, false, bl.v, (short)0, a1, false, false);
        a2 = __builtin_amdgcn_wmma_f32_16x16x32_bf16(false, al.v, false, bh.v, (short)0, a2, false, false);
    }
}

// ---------------- kernel 2: WMMA argmin over codebook ----------------
// block = 128 threads (4 waves), 64 tokens/block, each wave owns one 16-token tile.
// 512 code tiles; E tile staged via async-to-LDS, software-pipelined at K-half
// granularity inside a single 32KB (per plane 16KB) buffer:
//   [compute K<256 of ct  || async-load K>=256 of ct]
//   [compute K>=256 of ct || async-load K<256 of ct+1]
// score = z.e - 0.5*||e||^2 ; argmax(score) == argmin(dist).
__launch_bounds__(128)
__global__ void vq_argmin_wmma(const float* __restrict__ z,
                               const uint16_t* __restrict__ ehg,
                               const uint16_t* __restrict__ elg,
                               const float* __restrict__ hnorm,
                               int* __restrict__ idx_out,
                               float* __restrict__ idxf_out) {
    extern __shared__ char smem[];
    uint16_t* zh = (uint16_t*)smem;          // 64*512 halves = 64KB
    uint16_t* zl = zh + 64 * DIM;            // 64KB
    uint16_t* eh = zl + 64 * DIM;            // 16KB
    uint16_t* el = eh + 16 * DIM;            // 16KB
    const uint32_t eh_off = (uint32_t)(uintptr_t)eh;   // LDS byte address (addr[31:0])
    const uint32_t el_off = (uint32_t)(uintptr_t)el;

    const int tid  = threadIdx.x;
    const int lane = tid & 31;
    const int wave = tid >> 5;               // 0..3
    const int hlf  = lane >> 4;              // half-wave select
    const int mn   = lane & 15;              // A row (code) / B col (token) within tile

    const size_t t0 = (size_t)blockIdx.x * 64;
    const int tokrow = wave * 16 + mn;       // this lane's token row in LDS

    // stage z tile -> bf16 hi/lo in LDS (coalesced f32 reads)
    for (int i = tid; i < 64 * DIM; i += 128) {
        float f = z[t0 * DIM + i];
        uint16_t hi = f32_to_bf16_rne(f);
        zh[i] = hi;
        zl[i] = f32_to_bf16_rne(f - bf16_to_f32(hi));
    }

    // prologue: async-load region A (K<256) of tile 0
    copy_half_tile_async(ehg, eh_off, 0, 0, tid);
    copy_half_tile_async(elg, el_off, 0, 0, tid);
    wait_async0();
    __syncthreads();

    float best_s = -3.4e38f;
    int   best_i = 0;
    const int NT = K_CODES / 16;

    for (int ct = 0; ct < NT; ++ct) {
        // async-load region B (K>=256) of current tile while computing region A
        copy_half_tile_async(ehg, eh_off, ct, 256, tid);
        copy_half_tile_async(elg, el_off, ct, 256, tid);

        v8f a0, a1, a2;
        #pragma unroll
        for (int i = 0; i < 8; ++i) { a0[i] = 0.f; a1[i] = 0.f; a2[i] = 0.f; }

        compute_half(eh, el, zh, zl, mn, hlf, tokrow, 0, a0, a1, a2);

        wait_async0();
        __syncthreads();

        if (ct + 1 < NT) {
            // async-load region A of next tile while computing region B
            copy_half_tile_async(ehg, eh_off, ct + 1, 0, tid);
            copy_half_tile_async(elg, el_off, ct + 1, 0, tid);
            // warm GL2 for next tile's region B
            const char* nh = (const char*)(ehg + (size_t)(ct + 1) * 16 * DIM);
            const char* nl = (const char*)(elg + (size_t)(ct + 1) * 16 * DIM);
            __builtin_prefetch(nh + tid * 128, 0, 0);
            __builtin_prefetch(nl + tid * 128, 0, 0);
        }

        compute_half(eh, el, zh, zl, mn, hlf, tokrow, 256, a0, a1, a2);

        // D layout: VGPR r, lane => code m = r + 8*hlf, token n = mn
        const float* hp = hnorm + ct * 16 + 8 * hlf;
        #pragma unroll
        for (int r = 0; r < 8; ++r) {
            float sc = (a0[r] + a1[r] + a2[r]) - hp[r];
            int code = ct * 16 + 8 * hlf + r;
            if (sc > best_s) { best_s = sc; best_i = code; }
        }

        wait_async0();
        __syncthreads();
    }

    // merge the two half-wave code subsets for the same token
    float os = __shfl_xor(best_s, 16, 32);
    int   oi = __shfl_xor(best_i, 16, 32);
    if (os > best_s || (os == best_s && oi < best_i)) { best_s = os; best_i = oi; }
    if (hlf == 0) {
        size_t tok = t0 + (size_t)tokrow;
        idx_out[tok]  = best_i;
        idxf_out[tok] = (float)best_i;
    }
}

// ---------------- kernel 3: gather z_q (straight-through) + commitment loss ----------------
__global__ void vq_gather_loss(const float* __restrict__ z,
                               const float* __restrict__ emb,
                               const int* __restrict__ idx,
                               float* __restrict__ zq_out,
                               float* __restrict__ loss_out) {
    const int n = blockIdx.x;
    const int c = idx[n];
    const float* e  = emb + (size_t)c * DIM;
    const float* zr = z + (size_t)n * DIM;
    float acc = 0.f;
    for (int d = threadIdx.x; d < DIM; d += 256) {
        float q  = e[d];
        float zi = zr[d];
        zq_out[(size_t)n * DIM + d] = zi + (q - zi);  // straight-through estimator
        float diff = q - zi;
        acc += diff * diff;
    }
    for (int off = 16; off > 0; off >>= 1) acc += __shfl_down(acc, off, 32);
    __shared__ float red[8];
    if ((threadIdx.x & 31) == 0) red[threadIdx.x >> 5] = acc;
    __syncthreads();
    if (threadIdx.x == 0) {
        float s = 0.f;
        for (int i = 0; i < 8; ++i) s += red[i];
        atomicAdd(loss_out, s * (COMMIT / ((float)N_TOK * (float)DIM)));
    }
}

// ---------------- kernel 4: segment sums (counts + per-code embed sum) ----------------
__global__ void vq_scatter(const float* __restrict__ z, const int* __restrict__ idx,
                           float* __restrict__ counts, float* __restrict__ esum) {
    const int n = blockIdx.x;
    const int c = idx[n];
    if (threadIdx.x == 0) atomicAdd(counts + c, 1.0f);
    for (int d = threadIdx.x; d < DIM; d += 256)
        atomicAdd(esum + (size_t)c * DIM + d, z[(size_t)n * DIM + d]);
}

// ---------------- kernel 5: EMA cluster size + total n ----------------
__global__ void vq_ema_cs(const float* __restrict__ ema_cs_in, const float* __restrict__ counts,
                          float* __restrict__ cs_out, float* __restrict__ n_slot) {
    const int k = blockIdx.x * 256 + threadIdx.x;
    float v = 0.f;
    if (k < K_CODES) {
        v = ema_cs_in[k] * DECAY + counts[k] * (1.f - DECAY);
        cs_out[k] = v;
    }
    for (int off = 16; off > 0; off >>= 1) v += __shfl_down(v, off, 32);
    __shared__ float red[8];
    if ((threadIdx.x & 31) == 0) red[threadIdx.x >> 5] = v;
    __syncthreads();
    if (threadIdx.x == 0) {
        float s = 0.f;
        for (int i = 0; i < 8; ++i) s += red[i];
        atomicAdd(n_slot, s);
    }
}

// ---------------- kernel 6: EMA embed sum + normalized new embedding ----------------
__global__ void vq_final(const float* __restrict__ ema_es_in, const float* __restrict__ esum,
                         const float* __restrict__ cs_out, const float* __restrict__ n_slot,
                         float* __restrict__ es_out, float* __restrict__ newemb_out) {
    const size_t i = (size_t)blockIdx.x * 256 + threadIdx.x;
    if (i >= (size_t)K_CODES * DIM) return;
    const int k = (int)(i / DIM);
    float es = ema_es_in[i] * DECAY + esum[i] * (1.f - DECAY);
    es_out[i] = es;
    float n  = *n_slot;
    float cs = cs_out[k];
    float smoothed = (cs + EPSF) / (n + (float)K_CODES * EPSF) * n;
    newemb_out[i] = es / smoothed;
}

// ---------------- launcher ----------------
extern "C" void kernel_launch(void* const* d_in, const int* in_sizes, int n_in,
                              void* d_out, int out_size, void* d_ws, size_t ws_size,
                              hipStream_t stream) {
    const float* z      = (const float*)d_in[0];  // (8,2048,512)
    const float* emb    = (const float*)d_in[1];  // (8192,512)
    const float* ema_cs = (const float*)d_in[2];  // (8192,)
    const float* ema_es = (const float*)d_in[3];  // (8192,512)

    float* out = (float*)d_out;
    float* out_zq   = out + OFF_ZQ;
    float* out_loss = out + OFF_LOSS;
    float* out_idx  = out + OFF_IDX;
    float* out_nemb = out + OFF_NEWEMB;
    float* out_cs   = out + OFF_CS;
    float* out_es   = out + OFF_ES;

    // workspace carve-up
    char* ws = (char*)d_ws;
    uint16_t* eh   = (uint16_t*)ws;                               ws += (size_t)K_CODES * DIM * 2;   // 8MB
    uint16_t* el   = (uint16_t*)ws;                               ws += (size_t)K_CODES * DIM * 2;   // 8MB
    float* hnorm   = (float*)ws;                                  ws += (size_t)K_CODES * 4;         // 32KB
    int*   idx     = (int*)ws;                                    ws += (size_t)N_TOK * 4;           // 64KB
    float* counts  = (float*)ws;                                  ws += (size_t)K_CODES * 4;         // 32KB
    float* esum    = (float*)ws;                                  ws += (size_t)K_CODES * DIM * 4;   // 16MB
    float* n_slot  = (float*)ws;

    // 0: zero accumulators
    vq_init<<<(K_CODES * DIM + 255) / 256, 256, 0, stream>>>(counts, esum, out_loss, n_slot);
    // 1: codebook -> bf16 hi/lo + half norms
    vq_prep_codebook<<<K_CODES, 256, 0, stream>>>(emb, eh, el, hnorm);
    // 2: WMMA argmin (64 tokens/block, 160KB dynamic LDS -> 2 blocks per WGP)
    const size_t smem = (size_t)(64 * DIM * 2 + 16 * DIM * 2) * 2;  // 163840 B
    vq_argmin_wmma<<<N_TOK / 64, 128, smem, stream>>>(z, eh, el, hnorm, idx, out_idx);
    // 3: gather quantized vectors + loss
    vq_gather_loss<<<N_TOK, 256, 0, stream>>>(z, emb, idx, out_zq, out_loss);
    // 4: segment sums
    vq_scatter<<<N_TOK, 256, 0, stream>>>(z, idx, counts, esum);
    // 5: EMA cluster sizes + total n
    vq_ema_cs<<<(K_CODES + 255) / 256, 256, 0, stream>>>(ema_cs, counts, out_cs, n_slot);
    // 6: EMA embed sums + new embedding
    vq_final<<<(K_CODES * DIM + 255) / 256, 256, 0, stream>>>(ema_es, esum, out_cs, n_slot, out_es, out_nemb);
}
